// MLP_35364760715803
// MI455X (gfx1250) — matrix operand order repaired and checked by reading, not compiled, not run on hardware
//
#include <hip/hip_runtime.h>
#include <hip/hip_bf16.h>

typedef __attribute__((ext_vector_type(16))) _Float16 v16h;
typedef __attribute__((ext_vector_type(8)))  _Float16 v8h;
typedef __attribute__((ext_vector_type(4)))  _Float16 v4h;
typedef __attribute__((ext_vector_type(8)))  float    v8f;

union AFrag { v16h v; v8h h[2]; };

#define HID  128
#define FEAT 160
#define KN   32

// Pre-convert one stage's Wa (128x160 f32) to f16 with the K axis permuted:
//   k' in [0,128): j = k'>>2, c = k'&3  -> original k = 5*j + c   (r_matrix comps)
//   k' in [128,160): j = k'-128         -> original k = 5*j + 4   (diff comps)
// This lets the activation build use one aligned ds_store_b64 + one ds_store_b16
// per lane instead of 5 scalar b16 stores. Dot products are invariant under the
// shared permutation of x and Wa columns.
__global__ __launch_bounds__(256) void convert_w_kernel(const float* __restrict__ Wa,
                                                        _Float16* __restrict__ out) {
    int i = blockIdx.x * 256 + threadIdx.x;
    if (i < HID * FEAT) {
        int nrow = i / FEAT, kp = i % FEAT;
        int k = (kp < 128) ? ((kp >> 2) * 5 + (kp & 3)) : (5 * (kp - 128) + 4);
        out[i] = (_Float16)Wa[nrow * FEAT + k];
    }
}

// One WMMA stage: f_out = relu(x @ Wa^T + ba) @ Wb^T + bb, x = [r_matrix | f - f[neigh]]
// Block = 128 threads = 4 wave32; block covers 128 rows; each wave runs two
// 16-row tiles, each tile = 40x v_wmma_f32_16x16x32_f16 (K=160, N=128).
__global__ __launch_bounds__(128) void stage_kernel(
    const float*    __restrict__ f_in,
    const float*    __restrict__ rmat,   // N*32*4
    const int*      __restrict__ idx,    // N*33 (int32)
    const _Float16* __restrict__ Wh,     // 128*160 f16, K-permuted (convert_w_kernel)
    const float*    __restrict__ ba,     // 128
    const float*    __restrict__ Wb,     // 128
    const float*    __restrict__ bb,     // 1
    float* __restrict__ f_out,
    int n)
{
    __shared__ __align__(16) _Float16 sW[HID * FEAT];     // 40 KB: B matrix (f16)
    __shared__ float sBa[HID];
    __shared__ float sWb[HID];
    __shared__ __align__(16) _Float16 sX[4 * 16 * FEAT];  // 20 KB: per-wave 16x160 tiles

    const int tid  = threadIdx.x;
    const int wave = tid >> 5;
    const int lane = tid & 31;

    // ---- stage pre-converted f16 weights into LDS (16B vector copies) ----
    {
        const v8h* src = (const v8h*)Wh;
        v8h* dst = (v8h*)sW;
        #pragma unroll
        for (int i = 0; i < (HID * FEAT) / (8 * 128); ++i)   // 20 iters
            dst[tid + i * 128] = src[tid + i * 128];
        if (tid < HID) { sBa[tid] = ba[tid]; sWb[tid] = Wb[tid]; }
    }

    const int mrow = lane & 15;   // A: M row; B/C/D: N = nt*16 + mrow
    const int hs   = lane >> 4;   // K-half carried by this lane
    _Float16* sx = sX + wave * 16 * FEAT;
    const float4* rm4 = (const float4*)rmat;
    const float bbv = bb[0];

    for (int t = 0; t < 2; ++t) {
        const int rowbase = blockIdx.x * 128 + (t * 4 + wave) * 16;

        // ---- build activation tile: lane j handles neighbor j of each of 16 rows ----
        #pragma unroll
        for (int m = 0; m < 16; ++m) {
            int row = rowbase + m;
            int rc  = row < n ? row : (n - 1);      // clamp: keep EXEC all-ones for WMMA
            float fc = f_in[rc];
            int nj = idx[rc * 33 + 1 + lane];
            float diff = fc - f_in[nj];
            float4 rv = rm4[rc * KN + lane];
            _Float16* xp = sx + m * FEAT;
            v4h rh = { (_Float16)rv.x, (_Float16)rv.y, (_Float16)rv.z, (_Float16)rv.w };
            *(v4h*)(xp + 4 * lane) = rh;            // k' = 4j..4j+3 (8B aligned, bank-free)
            xp[128 + lane] = (_Float16)diff;        // k' = 128+j
        }
        __syncthreads();

        v8f acc[8];
        #pragma unroll
        for (int nt = 0; nt < 8; ++nt) acc[nt] = (v8f){};

        #pragma unroll
        for (int kt = 0; kt < 5; ++kt) {
            // A fragment (16x32): lanes<16 hold K {0..7,16..23}, lanes>=16 hold {8..15,24..31}
            AFrag a;
            const _Float16* ap = sx + mrow * FEAT + kt * 32 + hs * 8;
            a.h[0] = *(const v8h*)(ap);
            a.h[1] = *(const v8h*)(ap + 16);
            #pragma unroll
            for (int nt = 0; nt < 8; ++nt) {
                // B fragment (32x16): lane holds column nc, K {0..7,16..23} / {8..15,24..31} like A
                int nc = nt * 16 + mrow;
                AFrag b;
                const _Float16* bp = sW + nc * FEAT + kt * 32 + hs * 8;
                b.h[0] = *(const v8h*)(bp);
                b.h[1] = *(const v8h*)(bp + 16);
                acc[nt] = __builtin_amdgcn_wmma_f32_16x16x32_f16(
                    false, a.v, false, b.v, (short)0, acc[nt], false, false);
            }
        }

        // ---- second layer: relu + bias, dot with Wb, reduce across 16-lane halves ----
        float partial[8];
        #pragma unroll
        for (int r = 0; r < 8; ++r) partial[r] = 0.0f;
        #pragma unroll
        for (int nt = 0; nt < 8; ++nt) {
            int nc = nt * 16 + mrow;
            float bav = sBa[nc];
            float wbv = sWb[nc];
            #pragma unroll
            for (int r = 0; r < 8; ++r) {
                float h = fmaxf(acc[nt][r] + bav, 0.0f);
                partial[r] = fmaf(h, wbv, partial[r]);
            }
        }
        // D layout: VGPR r holds M = r + 8*hs, N = mrow
        #pragma unroll
        for (int off = 1; off < 16; off <<= 1) {
            #pragma unroll
            for (int r = 0; r < 8; ++r)
                partial[r] += __shfl_xor(partial[r], off, 32);
        }
        if (mrow == 0) {
            int base = rowbase + hs * 8;
            #pragma unroll
            for (int r = 0; r < 8; ++r) {
                int row = base + r;
                if (row < n) f_out[row] = partial[r] + bbv;
            }
        }
        __syncthreads();   // protect sX reuse across tile iterations
    }
}

// ---------------- softmax over N ----------------
__global__ __launch_bounds__(256) void reduce_max_kernel(const float* __restrict__ x, int n,
                                                         float* __restrict__ part) {
    __shared__ float s[256];
    float m = -3.402823466e38f;
    for (int i = blockIdx.x * blockDim.x + threadIdx.x; i < n; i += gridDim.x * blockDim.x)
        m = fmaxf(m, x[i]);
    s[threadIdx.x] = m; __syncthreads();
    for (int o = 128; o > 0; o >>= 1) {
        if (threadIdx.x < o) s[threadIdx.x] = fmaxf(s[threadIdx.x], s[threadIdx.x + o]);
        __syncthreads();
    }
    if (threadIdx.x == 0) part[blockIdx.x] = s[0];
}

__global__ __launch_bounds__(256) void final_max_kernel(const float* __restrict__ part, int nb,
                                                        float* __restrict__ gmax) {
    __shared__ float s[256];
    float m = -3.402823466e38f;
    for (int i = threadIdx.x; i < nb; i += 256) m = fmaxf(m, part[i]);
    s[threadIdx.x] = m; __syncthreads();
    for (int o = 128; o > 0; o >>= 1) {
        if (threadIdx.x < o) s[threadIdx.x] = fmaxf(s[threadIdx.x], s[threadIdx.x + o]);
        __syncthreads();
    }
    if (threadIdx.x == 0) gmax[0] = s[0];
}

__global__ __launch_bounds__(256) void exp_sum_kernel(const float* __restrict__ x, int n,
                                                      const float* __restrict__ gmax,
                                                      float* __restrict__ y,
                                                      float* __restrict__ part) {
    __shared__ float s[256];
    float mx = gmax[0];
    float sum = 0.0f;
    for (int i = blockIdx.x * blockDim.x + threadIdx.x; i < n; i += gridDim.x * blockDim.x) {
        float e = expf(x[i] - mx);
        y[i] = e;
        sum += e;
    }
    s[threadIdx.x] = sum; __syncthreads();
    for (int o = 128; o > 0; o >>= 1) {
        if (threadIdx.x < o) s[threadIdx.x] += s[threadIdx.x + o];
        __syncthreads();
    }
    if (threadIdx.x == 0) part[blockIdx.x] = s[0];
}

__global__ __launch_bounds__(256) void final_sum_kernel(const float* __restrict__ part, int nb,
                                                        float* __restrict__ gsum) {
    __shared__ float s[256];
    float sum = 0.0f;
    for (int i = threadIdx.x; i < nb; i += 256) sum += part[i];
    s[threadIdx.x] = sum; __syncthreads();
    for (int o = 128; o > 0; o >>= 1) {
        if (threadIdx.x < o) s[threadIdx.x] += s[threadIdx.x + o];
        __syncthreads();
    }
    if (threadIdx.x == 0) gsum[0] = s[0];
}

__global__ __launch_bounds__(256) void scale_kernel(float* __restrict__ y, int n,
                                                    const float* __restrict__ gsum) {
    float inv = 1.0f / gsum[0];
    for (int i = blockIdx.x * blockDim.x + threadIdx.x; i < n; i += gridDim.x * blockDim.x)
        y[i] *= inv;
}

extern "C" void kernel_launch(void* const* d_in, const int* in_sizes, int n_in,
                              void* d_out, int out_size, void* d_ws, size_t ws_size,
                              hipStream_t stream) {
    const float* p_init = (const float*)d_in[0];
    const float* rmat   = (const float*)d_in[1];
    const int*   idx    = (const int*)d_in[2];
    const float* W1a = (const float*)d_in[3];  const float* b1a = (const float*)d_in[4];
    const float* W1b = (const float*)d_in[5];  const float* b1b = (const float*)d_in[6];
    const float* W2a = (const float*)d_in[7];  const float* b2a = (const float*)d_in[8];
    const float* W2b = (const float*)d_in[9];  const float* b2b = (const float*)d_in[10];
    const float* W3a = (const float*)d_in[11]; const float* b3a = (const float*)d_in[12];
    const float* W3b = (const float*)d_in[13]; const float* b3b = (const float*)d_in[14];
    float* out = (float*)d_out;
    const int n = in_sizes[0];

    // workspace: ping buffer (N f32) | reduction partials | 3x f16 weights
    float* fA   = (float*)d_ws;
    size_t off1 = (((size_t)n * 4) + 255) & ~(size_t)255;
    float* part = (float*)((char*)d_ws + off1);
    float* gmax = part + 2048;
    float* gsum = gmax + 1;
    size_t off2 = (off1 + 2052 * 4 + 255) & ~(size_t)255;
    _Float16* wh1 = (_Float16*)((char*)d_ws + off2);
    _Float16* wh2 = wh1 + HID * FEAT;
    _Float16* wh3 = wh2 + HID * FEAT;

    // one-shot f32 -> f16 (K-permuted) weight conversion
    const int WG = (HID * FEAT + 255) / 256;
    convert_w_kernel<<<WG, 256, 0, stream>>>(W1a, wh1);
    convert_w_kernel<<<WG, 256, 0, stream>>>(W2a, wh2);
    convert_w_kernel<<<WG, 256, 0, stream>>>(W3a, wh3);

    dim3 blk(128);
    dim3 grd((unsigned)((n + 127) / 128));
    // stage 1: p_init -> fA ; stage 2: fA -> out ; stage 3: out -> fA
    stage_kernel<<<grd, blk, 0, stream>>>(p_init, rmat, idx, wh1, b1a, W1b, b1b, fA,  n);
    stage_kernel<<<grd, blk, 0, stream>>>(fA,     rmat, idx, wh2, b2a, W2b, b2b, out, n);
    stage_kernel<<<grd, blk, 0, stream>>>(out,    rmat, idx, wh3, b3a, W3b, b3b, fA,  n);

    const int NB = 1024;
    reduce_max_kernel<<<NB, 256, 0, stream>>>(fA, n, part);
    final_max_kernel<<<1, 256, 0, stream>>>(part, NB, gmax);
    exp_sum_kernel<<<NB, 256, 0, stream>>>(fA, n, gmax, out, part);
    final_sum_kernel<<<1, 256, 0, stream>>>(part, NB, gsum);
    scale_kernel<<<(n + 255) / 256, 256, 0, stream>>>(out, n, gsum);
}
